// LoRAAttention_58626303590933
// MI455X (gfx1250) — matrix operand order repaired and checked by reading, not compiled, not run on hardware
//
#include <hip/hip_runtime.h>
#include <hip/hip_bf16.h>

typedef __attribute__((ext_vector_type(16))) _Float16 v16h;
typedef __attribute__((ext_vector_type(8)))  _Float16 v8h;
typedef __attribute__((ext_vector_type(8)))  float    v8f;
typedef int v4i __attribute__((__vector_size__(16)));

#define N_TOK   2304
#define DIM     768
#define HEADS   12
#define HD      64
#define HS      48
#define RANK    4
#define LSCALE  0.25f     /* ALPHA / RANK */
#define QSCALE  0.125f    /* HEAD_DIM ** -0.5 */

/* ---- CDNA5 async global->LDS path (guarded; falls back to uint4 + ds_store) ---- */
#if defined(__gfx1250__) && __has_builtin(__builtin_amdgcn_global_load_async_to_lds_b128)
#define USE_ASYNC_LDS 1
#else
#define USE_ASYNC_LDS 0
#endif

static __device__ __forceinline__ void async_copy16(const _Float16* g, _Float16* l) {
#if USE_ASYNC_LDS
    __builtin_amdgcn_global_load_async_to_lds_b128(
        (__attribute__((address_space(1))) v4i*)g,
        (__attribute__((address_space(3))) v4i*)l, 0, 0);
#else
    *(uint4*)l = *(const uint4*)g;
#endif
}
static __device__ __forceinline__ void async_wait0() {
#if USE_ASYNC_LDS
#if __has_builtin(__builtin_amdgcn_s_wait_asynccnt)
    __builtin_amdgcn_s_wait_asynccnt(0);
#else
    asm volatile("s_wait_asynccnt 0" ::: "memory");
#endif
#endif
}

static __device__ __forceinline__ v16h cat16(v8h lo, v8h hi) {
    return __builtin_shufflevector(lo, hi, 0,1,2,3,4,5,6,7,8,9,10,11,12,13,14,15);
}
// A-fragment style: 8 halves at p, 8 halves at p+16 (K {0..7} U {16..23} rel. base)
static __device__ __forceinline__ v16h ld_fragA(const _Float16* p) {
    return cat16(*(const v8h*)p, *(const v8h*)(p + 16));
}
// B-fragment style: 8 halves at p, 8 halves at p+16 (K {0..7} U {16..23} rel. base)
static __device__ __forceinline__ v16h ld_fragB(const _Float16* p) {
    return cat16(*(const v8h*)p, *(const v8h*)(p + 16));
}
static __device__ __forceinline__ v8f zero8() {
    v8f z;
#pragma unroll
    for (int i = 0; i < 8; ++i) z[i] = 0.0f;
    return z;
}

/* ---------------- elementwise f32 -> f16 cast ---------------- */
__global__ void k_f2h(const float* __restrict__ in, _Float16* __restrict__ out, int n) {
    int i = blockIdx.x * 256 + threadIdx.x;
    if (i < n) out[i] = (_Float16)in[i];
}

/* ---------------- xa[sel][row][r] = dot(x[row], A_sel[r]) ---------------- */
__global__ void k_lora_xa(const float* __restrict__ x,
                          const float* __restrict__ Aq,
                          const float* __restrict__ Ak,
                          const float* __restrict__ Av,
                          float* __restrict__ xa) {
    int i = blockIdx.x * 256 + threadIdx.x;
    if (i >= 3 * N_TOK * RANK) return;
    int r   = i & 3;
    int row = (i >> 2) % N_TOK;
    int sel = i / (N_TOK * RANK);
    const float* A = (sel == 0) ? Aq : (sel == 1) ? Ak : Av;
    const float4* xv = (const float4*)(x + (size_t)row * DIM);
    const float4* av = (const float4*)(A + (size_t)r * DIM);
    float s = 0.0f;
    for (int t = 0; t < DIM / 4; ++t) {
        float4 a = xv[t], b = av[t];
        s += a.x * b.x + a.y * b.y + a.z * b.z + a.w * b.w;
    }
    xa[i] = s;
}

/* ---------------- WMMA GEMM: C[M x Ncols] = A[M x 768] * W[Ncols x 768]^T ----------
   Double-buffered LDS pipeline: stage chunk i+1 (async) while computing chunk i.
   mode 0: QKV epilogue (bias + LoRA, scatter to qf/qh/kh/vh)
   mode 1: proj epilogue (bias, write f32 out)                              */
__global__ __launch_bounds__(128) void k_gemm(
    const _Float16* __restrict__ A, const _Float16* __restrict__ Bw,
    const float* __restrict__ bias, int mode,
    const float* __restrict__ xa,
    const float* __restrict__ lbq, const float* __restrict__ lbk, const float* __restrict__ lbv,
    float* __restrict__ qf, _Float16* __restrict__ qh,
    _Float16* __restrict__ kh, _Float16* __restrict__ vh,
    float* __restrict__ outF)
{
    __shared__ _Float16 As[2][64 * 32];
    __shared__ _Float16 Bs[2][64 * 32];
    const int tid   = threadIdx.x;
    const int w     = tid >> 5;
    const int lane  = tid & 31;
    const int lhalf = (lane >> 4) & 1;
    const int l16   = lane & 15;
    const int m0    = blockIdx.y * 64;
    const int n0    = blockIdx.x * 64;
    const int srow  = tid >> 1;
    const int sseg  = (tid & 1) * 16;

    v8f acc[4];
#pragma unroll
    for (int t = 0; t < 4; ++t) acc[t] = zero8();

    auto stage = [&](int kk, int buf) {
        const _Float16* ga = &A [(size_t)(m0 + srow) * DIM + kk + sseg];
        const _Float16* gb = &Bw[(size_t)(n0 + srow) * DIM + kk + sseg];
        async_copy16(ga,     &As[buf][srow * 32 + sseg]);
        async_copy16(ga + 8, &As[buf][srow * 32 + sseg + 8]);
        async_copy16(gb,     &Bs[buf][srow * 32 + sseg]);
        async_copy16(gb + 8, &Bs[buf][srow * 32 + sseg + 8]);
        if (kk + 32 < DIM) {
            __builtin_prefetch((const void*)(ga + 32), 0, 3);
            __builtin_prefetch((const void*)(gb + 32), 0, 3);
        }
    };

    stage(0, 0);
    async_wait0();
    __syncthreads();

    const int NSTEP = DIM / 32;
    for (int i = 0; i < NSTEP; ++i) {
        const int cur = i & 1;
        if (i + 1 < NSTEP) stage((i + 1) * 32, cur ^ 1);   // overlap with compute

        v16h afrag = ld_fragA(&As[cur][(16 * w + l16) * 32 + lhalf * 8]);
#pragma unroll
        for (int t = 0; t < 4; ++t) {
            v16h bfrag = ld_fragB(&Bs[cur][(16 * t + l16) * 32 + lhalf * 8]);
            acc[t] = __builtin_amdgcn_wmma_f32_16x16x32_f16(
                false, afrag, false, bfrag, (short)0, acc[t], false, false);
        }
        async_wait0();
        __syncthreads();
    }

#pragma unroll
    for (int t = 0; t < 4; ++t) {
        int col = n0 + 16 * t + l16;
        float b = bias[col];
#pragma unroll
        for (int r = 0; r < 8; ++r) {
            int row = m0 + 16 * w + r + 8 * lhalf;
            float val = acc[t][r] + b;
            if (mode == 0) {
                int sel = col / DIM;
                int ch  = col - sel * DIM;
                const float* Bl = ((sel == 0) ? lbq : (sel == 1) ? lbk : lbv) + ch * RANK;
                const float* xp = xa + ((size_t)sel * N_TOK + row) * RANK;
                val += LSCALE * (xp[0] * Bl[0] + xp[1] * Bl[1] + xp[2] * Bl[2] + xp[3] * Bl[3]);
                int head = ch >> 6, c = ch & 63;
                size_t oi = ((size_t)head * N_TOK + row) * HD + c;
                if (sel == 0) { qf[oi] = val; qh[oi] = (_Float16)(val * QSCALE); }
                else if (sel == 1) { kh[oi] = (_Float16)val; }
                else               { vh[oi] = (_Float16)val; }
            } else {
                outF[(size_t)row * DIM + col] = val;
            }
        }
    }
}

/* ---------------- decomposed relative position bias ---------------- */
__global__ void k_relbias(const float* __restrict__ qf,
                          const float* __restrict__ rph, const float* __restrict__ rpw,
                          float* __restrict__ relh, float* __restrict__ relw) {
    int i = blockIdx.x * 256 + threadIdx.x;
    if (i >= HEADS * N_TOK * HS) return;
    int kk   = i % HS;
    int qi   = (i / HS) % N_TOK;
    int head = i / (HS * N_TOK);
    int hq = qi / HS, wq = qi % HS;
    const float4* qv = (const float4*)(qf + ((size_t)head * N_TOK + qi) * HD);
    const float4* hv = (const float4*)(rph + (size_t)(hq - kk + HS - 1) * HD);
    const float4* wv = (const float4*)(rpw + (size_t)(wq - kk + HS - 1) * HD);
    float sh = 0.0f, sw = 0.0f;
    for (int t = 0; t < HD / 4; ++t) {
        float4 q = qv[t], a = hv[t], b = wv[t];
        sh += q.x * a.x + q.y * a.y + q.z * a.z + q.w * a.w;
        sw += q.x * b.x + q.y * b.y + q.z * b.z + q.w * b.w;
    }
    relh[i] = sh;
    relw[i] = sw;
}

/* ---------------- fused flash attention (per head, 64-query tile) ---------------- */
__global__ __launch_bounds__(128) void k_attn(
    const _Float16* __restrict__ qh, const _Float16* __restrict__ kh,
    const _Float16* __restrict__ vh,
    const float* __restrict__ relh, const float* __restrict__ relw,
    _Float16* __restrict__ oh)
{
    __shared__ _Float16 Ks[32 * 64];     // [key][ch]
    __shared__ _Float16 Vt[64 * 32];     // [ch][key]
    __shared__ _Float16 Ps[4][16 * 32];  // per-wave P repack

    const int tid   = threadIdx.x;
    const int w     = tid >> 5;
    const int lane  = tid & 31;
    const int lhalf = (lane >> 4) & 1;
    const int l16   = lane & 15;
    const int head  = blockIdx.y;
    const int qbase = blockIdx.x * 64 + 16 * w;
    const int crow  = tid >> 2;
    const int cseg  = (tid & 3) * 16;

    v16h qa[2];
#pragma unroll
    for (int s = 0; s < 2; ++s)
        qa[s] = ld_fragA(&qh[((size_t)head * N_TOK + qbase + l16) * HD + s * 32 + lhalf * 8]);

    v8f o[4];
#pragma unroll
    for (int t = 0; t < 4; ++t) o[t] = zero8();
    float mi[8], li[8];
#pragma unroll
    for (int r = 0; r < 8; ++r) { mi[r] = -1e30f; li[r] = 0.0f; }

    for (int c = 0; c < N_TOK; c += 32) {
        __syncthreads();
        {   // stage K (row-major, async path) and V (transposed, manual) for this chunk
            const _Float16* kg = &kh[((size_t)head * N_TOK + c + crow) * HD + cseg];
            async_copy16(kg,     &Ks[crow * 64 + cseg]);
            async_copy16(kg + 8, &Ks[crow * 64 + cseg + 8]);
            const _Float16* vg = &vh[((size_t)head * N_TOK + c + crow) * HD + cseg];
            v8h v0 = *(const v8h*)vg;
            v8h v1 = *(const v8h*)(vg + 8);
#pragma unroll
            for (int i2 = 0; i2 < 8; ++i2) {
                Vt[(cseg + i2) * 32 + crow]     = v0[i2];
                Vt[(cseg + 8 + i2) * 32 + crow] = v1[i2];
            }
            if (c + 32 < N_TOK)
                __builtin_prefetch((const void*)(kg + 32 * HD), 0, 3);
        }
        async_wait0();
        __syncthreads();

        // S = (q*scale) @ K^T : two 16x16 tiles, K-dim 64 in two steps of 32
        v8f st[2];
#pragma unroll
        for (int t = 0; t < 2; ++t) {
            st[t] = zero8();
#pragma unroll
            for (int s = 0; s < 2; ++s) {
                v16h bf = ld_fragB(&Ks[(16 * t + l16) * 64 + s * 32 + lhalf * 8]);
                st[t] = __builtin_amdgcn_wmma_f32_16x16x32_f16(
                    false, qa[s], false, bf, (short)0, st[t], false, false);
            }
        }
        // decomposed rel-pos bias
#pragma unroll
        for (int t = 0; t < 2; ++t) {
            int kj = c + 16 * t + l16;
            int hk = kj / HS, wk = kj - hk * HS;
#pragma unroll
            for (int r = 0; r < 8; ++r) {
                int qi = qbase + r + 8 * lhalf;
                st[t][r] += relh[((size_t)head * N_TOK + qi) * HS + hk]
                          + relw[((size_t)head * N_TOK + qi) * HS + wk];
            }
        }
        // online softmax over this 32-key chunk (row reduce across 16 lanes)
#pragma unroll
        for (int r = 0; r < 8; ++r) {
            float vmax = fmaxf(st[0][r], st[1][r]);
            vmax = fmaxf(vmax, __shfl_xor(vmax, 1, 32));
            vmax = fmaxf(vmax, __shfl_xor(vmax, 2, 32));
            vmax = fmaxf(vmax, __shfl_xor(vmax, 4, 32));
            vmax = fmaxf(vmax, __shfl_xor(vmax, 8, 32));
            float mnew = fmaxf(mi[r], vmax);
            float fs = __expf(mi[r] - mnew);
            mi[r] = mnew;
            float p0 = __expf(st[0][r] - mnew);
            float p1 = __expf(st[1][r] - mnew);
            st[0][r] = p0; st[1][r] = p1;
            float ps = p0 + p1;
            ps += __shfl_xor(ps, 1, 32);
            ps += __shfl_xor(ps, 2, 32);
            ps += __shfl_xor(ps, 4, 32);
            ps += __shfl_xor(ps, 8, 32);
            li[r] = li[r] * fs + ps;
#pragma unroll
            for (int t4 = 0; t4 < 4; ++t4) o[t4][r] *= fs;
        }
        // repack P (C-layout f32) -> A-fragment (f16) via per-wave LDS
#pragma unroll
        for (int t = 0; t < 2; ++t)
#pragma unroll
            for (int r = 0; r < 8; ++r)
                Ps[w][(r + 8 * lhalf) * 32 + 16 * t + l16] = (_Float16)st[t][r];
        v16h pa = ld_fragA(&Ps[w][l16 * 32 + lhalf * 8]);
        // O += P @ V
#pragma unroll
        for (int t4 = 0; t4 < 4; ++t4) {
            v16h bf = ld_fragB(&Vt[(16 * t4 + l16) * 32 + lhalf * 8]);
            o[t4] = __builtin_amdgcn_wmma_f32_16x16x32_f16(
                false, pa, false, bf, (short)0, o[t4], false, false);
        }
    }

#pragma unroll
    for (int t4 = 0; t4 < 4; ++t4)
#pragma unroll
        for (int r = 0; r < 8; ++r) {
            int qi = qbase + r + 8 * lhalf;
            int cc = 16 * t4 + l16;
            oh[(size_t)qi * DIM + head * HD + cc] = (_Float16)(o[t4][r] / li[r]);
        }
}

/* ---------------- host launch ---------------- */
extern "C" void kernel_launch(void* const* d_in, const int* in_sizes, int n_in,
                              void* d_out, int out_size, void* d_ws, size_t ws_size,
                              hipStream_t stream) {
    (void)in_sizes; (void)n_in; (void)out_size; (void)ws_size;
    const float* x      = (const float*)d_in[0];
    const float* qkv_w  = (const float*)d_in[1];
    const float* qkv_b  = (const float*)d_in[2];
    const float* proj_w = (const float*)d_in[3];
    const float* proj_b = (const float*)d_in[4];
    const float* rph    = (const float*)d_in[5];
    const float* rpw    = (const float*)d_in[6];
    const float* laq    = (const float*)d_in[7];
    const float* lbq    = (const float*)d_in[8];
    const float* lak    = (const float*)d_in[9];
    const float* lbk    = (const float*)d_in[10];
    const float* lav    = (const float*)d_in[11];
    const float* lbv    = (const float*)d_in[12];
    float* out = (float*)d_out;

    char* ws = (char*)d_ws;
    size_t off = 0;
    auto alloc = [&](size_t bytes) -> char* {
        char* p = ws + off;
        off = (off + bytes + 255) & ~(size_t)255;
        return p;
    };
    _Float16* xh     = (_Float16*)alloc((size_t)N_TOK * DIM * 2);
    _Float16* wqkvh  = (_Float16*)alloc((size_t)3 * DIM * DIM * 2);
    _Float16* wprojh = (_Float16*)alloc((size_t)DIM * DIM * 2);
    float*    xa     = (float*)   alloc((size_t)3 * N_TOK * RANK * 4);
    float*    qf     = (float*)   alloc((size_t)HEADS * N_TOK * HD * 4);
    _Float16* qh     = (_Float16*)alloc((size_t)HEADS * N_TOK * HD * 2);
    _Float16* kh     = (_Float16*)alloc((size_t)HEADS * N_TOK * HD * 2);
    _Float16* vh     = (_Float16*)alloc((size_t)HEADS * N_TOK * HD * 2);
    float*    relh   = (float*)   alloc((size_t)HEADS * N_TOK * HS * 4);
    float*    relw   = (float*)   alloc((size_t)HEADS * N_TOK * HS * 4);
    _Float16* oh     = (_Float16*)alloc((size_t)N_TOK * DIM * 2);

    k_f2h<<<(N_TOK * DIM) / 256, 256, 0, stream>>>(x, xh, N_TOK * DIM);
    k_f2h<<<(3 * DIM * DIM) / 256, 256, 0, stream>>>(qkv_w, wqkvh, 3 * DIM * DIM);
    k_f2h<<<(DIM * DIM) / 256, 256, 0, stream>>>(proj_w, wprojh, DIM * DIM);
    k_lora_xa<<<(3 * N_TOK * RANK) / 256, 256, 0, stream>>>(x, laq, lak, lav, xa);

    dim3 gQKV(3 * DIM / 64, N_TOK / 64);
    k_gemm<<<gQKV, 128, 0, stream>>>(xh, wqkvh, qkv_b, 0, xa, lbq, lbk, lbv,
                                     qf, qh, kh, vh, nullptr);

    k_relbias<<<(HEADS * N_TOK * HS) / 256, 256, 0, stream>>>(qf, rph, rpw, relh, relw);

    dim3 gATT(N_TOK / 64, HEADS);
    k_attn<<<gATT, 128, 0, stream>>>(qh, kh, vh, relh, relw, oh);

    dim3 gPRJ(DIM / 64, N_TOK / 64);
    k_gemm<<<gPRJ, 128, 0, stream>>>(oh, wprojh, proj_b, 1, nullptr, nullptr, nullptr, nullptr,
                                     nullptr, nullptr, nullptr, nullptr, out);
}
